// scan_clsNet_71339406786602
// MI455X (gfx1250) — matrix operand order repaired and checked by reading, not compiled, not run on hardware
//
#include <hip/hip_runtime.h>
#include <hip/hip_bf16.h>

#define B_  16
#define N_  2048
#define K_  20
#define C_  64
#define OUT_ 64
#define NEG_SLOPE 0.2f

typedef _Float16 f16;
typedef _Float16 v16h __attribute__((ext_vector_type(16)));
typedef float    v8f  __attribute__((ext_vector_type(8)));

union FragH { v16h v; unsigned u[8]; };
union FragF { v8f  v; float    f[8]; };

// ---- params layout in workspace float block ----
#define P_DKW1 0      // 16x8
#define P_DKB1 128    // 16
#define P_DKW2 144    // 32x16
#define P_DKB2 656    // 32
#define P_FFB1 688    // 32
#define P_EDGEB 720   // 64
#define P_ACTG 784    // 64
#define P_ACTB 848    // 64
#define P_TOT  912

// ===================== KNN =====================
__global__ __launch_bounds__(256) void knn_kernel(const float* __restrict__ pos,
                                                  int* __restrict__ idx) {
  const int b = blockIdx.y;
  const int n = blockIdx.x * 256 + threadIdx.x;
  const float* pb = pos + (size_t)b * 3 * N_;
  const float qx = pb[n], qy = pb[N_ + n], qz = pb[2 * N_ + n];

  float vals[K_];
  int ids[K_];
#pragma unroll
  for (int i = 0; i < K_; ++i) { vals[i] = -3.0e38f; ids[i] = 0; }

  __shared__ float sx[256], sy[256], sz[256];
  for (int m0 = 0; m0 < N_; m0 += 256) {
    __syncthreads();
    sx[threadIdx.x] = pb[m0 + threadIdx.x];
    sy[threadIdx.x] = pb[N_ + m0 + threadIdx.x];
    sz[threadIdx.x] = pb[2 * N_ + m0 + threadIdx.x];
    __syncthreads();
    for (int mm = 0; mm < 256; ++mm) {
      const int m = m0 + mm;
      const float dx = qx - sx[mm], dy = qy - sy[mm], dz = qz - sz[mm];
      const float d = -(dx * dx + dy * dy + dz * dz);  // pd, maximize
      if (m != n && d > vals[K_ - 1]) {
        vals[K_ - 1] = d; ids[K_ - 1] = m;
#pragma unroll
        for (int j = K_ - 1; j > 0; --j) {
          if (vals[j] > vals[j - 1]) {
            float tv = vals[j]; vals[j] = vals[j - 1]; vals[j - 1] = tv;
            int   ti = ids[j];  ids[j]  = ids[j - 1];  ids[j - 1]  = ti;
          }
        }
      }
    }
  }
  int* op = idx + ((size_t)(b * N_ + n)) * K_;
#pragma unroll
  for (int i = 0; i < K_; ++i) op[i] = ids[i];
}

// ===================== bx = basis_W @ x =====================
__global__ __launch_bounds__(256) void bx_kernel(const float* __restrict__ x,
                                                 const float* __restrict__ W,
                                                 float* __restrict__ bx) {
  const int b = blockIdx.y;
  const int n0 = blockIdx.x * 64;
  __shared__ float Ws[64 * 64];
  __shared__ float xs[64][65];
  for (int i = threadIdx.x; i < 64 * 64; i += 256) Ws[i] = W[i];
  for (int i = threadIdx.x; i < 64 * 64; i += 256) {
    const int c = i >> 6, nn = i & 63;
    xs[c][nn] = x[((size_t)b * 64 + c) * N_ + n0 + nn];
  }
  __syncthreads();
  for (int i = threadIdx.x; i < 64 * 64; i += 256) {
    const int o = i >> 6, nn = i & 63;
    float acc = 0.f;
#pragma unroll
    for (int c = 0; c < 64; ++c) acc += Ws[o * 64 + c] * xs[c][nn];
    bx[((size_t)b * 64 + o) * N_ + n0 + nn] = acc;
  }
}

// ===================== weight prep (fold BN gains, cast to f16) =====================
__global__ void prep_kernel(const float* dk_W1, const float* dk_b1, const float* dk_g1,
                            const float* dk_be1, const float* dk_W2, const float* dk_b2,
                            const float* act_g, const float* act_b,
                            const float* ff_W1, const float* ff_g1, const float* ff_be1,
                            const float* ff_W2,
                            const float* edge_W, const float* edge_g, const float* edge_be,
                            f16* ffW1h, f16* ffW2h, f16* edgeWh, float* params) {
  const int t = threadIdx.x;
  for (int i = t; i < 32 * 64; i += 512) { int o = i >> 6; ffW1h[i] = (f16)(ff_g1[o] * ff_W1[i]); }
  for (int i = t; i < 64 * 32; i += 512) ffW2h[i] = (f16)ff_W2[i];
  for (int i = t; i < 64 * 128; i += 512) { int o = i >> 7; edgeWh[i] = (f16)(edge_g[o] * edge_W[i]); }
  for (int i = t; i < 16 * 8; i += 512) { int o = i >> 3; params[P_DKW1 + i] = dk_g1[o] * dk_W1[i]; }
  for (int i = t; i < 16; i += 512) params[P_DKB1 + i] = dk_g1[i] * dk_b1[i] + dk_be1[i];
  for (int i = t; i < 32 * 16; i += 512) params[P_DKW2 + i] = dk_W2[i];
  for (int i = t; i < 32; i += 512) params[P_DKB2 + i] = dk_b2[i];
  for (int i = t; i < 32; i += 512) params[P_FFB1 + i] = ff_be1[i];
  for (int i = t; i < 64; i += 512) params[P_EDGEB + i] = edge_be[i];
  for (int i = t; i < 64; i += 512) params[P_ACTG + i] = act_g[i];
  for (int i = t; i < 64; i += 512) params[P_ACTB + i] = act_b[i];
}

// ===================== fused main kernel =====================
#define NPTS 8
#define PIX  (NPTS * K_)   // 160 pixels, 160 threads, 5 waves
#define SROW 136           // padded LDS row stride (halfs) to dodge bank conflicts
#define NWAVE (PIX / 32)   // 5

__device__ __forceinline__ unsigned pack2(float a, float b) {
  union { f16 h; unsigned short s; } ua, ub;
  ua.h = (f16)a; ub.h = (f16)b;
  return (unsigned)ua.s | ((unsigned)ub.s << 16);
}

// A-fragment (16x32 f16, M x K) from row-major f16 weights W[O][Cdim].
// lane row M = lane%16; lanes<16 carry K in {0..15}(+base),
// lanes>=16 carry K in {16..31}(+base); dword j packs two halfs.
__device__ __forceinline__ v16h load_a_frag(const f16* __restrict__ W, int Cdim,
                                            int mt, int ks, int lr, int l2) {
  FragH f;
  const unsigned* base =
      (const unsigned*)(W + (size_t)(mt * 16 + lr) * Cdim + ks * 32 + 16 * l2);
#pragma unroll
  for (int j = 0; j < 8; ++j) f.u[j] = base[j];
  return f.v;
}

// B-fragment (32x16 f16, K x N) from LDS tile buf[p][c] (row stride SROW halfs).
// lanes 0-15: N=lane, K=0..15; lanes 16-31: N=lane-16, K=16..31.
__device__ __forceinline__ v16h load_b_frag(const f16* bufp, int p0, int c0,
                                            int lr, int l2) {
  FragH f;
  const unsigned* base = (const unsigned*)(bufp + (p0 + lr) * SROW + c0 + 16 * l2);
#pragma unroll
  for (int j = 0; j < 8; ++j) f.u[j] = base[j];
  return f.v;
}

__global__ __launch_bounds__(PIX) void fused_kernel(
    const float* __restrict__ x, const float* __restrict__ glf,
    const float* __restrict__ appf, const int* __restrict__ idx,
    const float* __restrict__ bx,
    const f16* __restrict__ ffW1h, const f16* __restrict__ ffW2h,
    const f16* __restrict__ edgeWh, const float* __restrict__ paramsg,
    float* __restrict__ out) {
  __shared__ f16 buf[PIX * SROW];       // 43.5 KB staging tile
  __shared__ float prm[P_TOT];
  __shared__ int outi[NPTS * OUT_];

  const int tid = threadIdx.x;
  const int b = blockIdx.y;
  const int n0 = blockIdx.x * NPTS;

  for (int i = tid; i < P_TOT; i += PIX) prm[i] = paramsg[i];
  for (int i = tid; i < NPTS * OUT_; i += PIX) outi[i] = (int)0x80000000;
  __syncthreads();

  // ---------- Phase 1: per-pixel dk MLP -> rif[64], stage f16 to LDS ----------
  const int p = tid;
  const int pt = p / K_;
  const int kk = p - pt * K_;
  const int n = n0 + pt;
  const int nbr = idx[((size_t)(b * N_ + n)) * K_ + kk];

  float ap[8];
  {
    const float* aptr = appf + (((size_t)(b * N_ + n)) * K_ + kk) * 8;
#pragma unroll
    for (int j = 0; j < 8; ++j) ap[j] = aptr[j];
  }
  float h[16];
#pragma unroll
  for (int i = 0; i < 16; ++i) {
    float s = prm[P_DKB1 + i];
#pragma unroll
    for (int j = 0; j < 8; ++j) s += prm[P_DKW1 + i * 8 + j] * ap[j];
    h[i] = fmaxf(s, 0.f);
  }
  float rif[64];
#pragma unroll
  for (int o = 0; o < 32; ++o) {
    float s = prm[P_DKB2 + o];
#pragma unroll
    for (int i = 0; i < 16; ++i) s += prm[P_DKW2 + o * 16 + i] * h[i];
    rif[o] = s;
  }
  {
    const float* gptr = glf + (((size_t)b * 32) * N_ + n) * K_ + kk;
#pragma unroll
    for (int c = 0; c < 32; ++c) rif[32 + c] = gptr[(size_t)c * (N_ * K_)];
  }
#pragma unroll
  for (int c = 0; c < 64; c += 2)
    *(unsigned*)&buf[p * SROW + c] = pack2(rif[c], rif[c + 1]);
  __syncthreads();

  const int lane = tid & 31;
  const int wv = tid >> 5;
  const int lr = lane & 15;
  const int l2 = lane >> 4;

  // ---------- GEMM1 (WMMA): att1(32 x PIX) = ffW1h(32x64) * rif; -> c 64..95 ----------
  for (int nt = wv * 2; nt < wv * 2 + 2; ++nt) {
    const int p0 = nt * 16;
    v8f acc0 = {}, acc1 = {};
#pragma unroll
    for (int ks = 0; ks < 2; ++ks) {
      v16h bf = load_b_frag(buf, p0, ks * 32, lr, l2);
      v16h a0 = load_a_frag(ffW1h, 64, 0, ks, lr, l2);
      acc0 = __builtin_amdgcn_wmma_f32_16x16x32_f16(false, a0, false, bf, (short)0, acc0, false, false);
      v16h a1 = load_a_frag(ffW1h, 64, 1, ks, lr, l2);
      acc1 = __builtin_amdgcn_wmma_f32_16x16x32_f16(false, a1, false, bf, (short)0, acc1, false, false);
    }
#pragma unroll
    for (int mt = 0; mt < 2; ++mt) {
      FragF ff; ff.v = mt ? acc1 : acc0;
#pragma unroll
      for (int i = 0; i < 8; i += 2) {
        const int o = mt * 16 + 8 * l2 + i;
        float v0 = ff.f[i] + prm[P_FFB1 + o];
        float v1 = ff.f[i + 1] + prm[P_FFB1 + o + 1];
        v0 = v0 > 0.f ? v0 : NEG_SLOPE * v0;
        v1 = v1 > 0.f ? v1 : NEG_SLOPE * v1;
        *(unsigned*)&buf[(p0 + lr) * SROW + 64 + o] = pack2(v0, v1);
      }
    }
  }
  __syncthreads();

  // ---------- GEMM2 (WMMA): att(64 x PIX) = ffW2h(64x32) * att1; sigmoid -> c 0..63 ----------
  for (int nt = wv * 2; nt < wv * 2 + 2; ++nt) {
    const int p0 = nt * 16;
    v16h bf = load_b_frag(buf, p0, 64, lr, l2);
    v8f acc[4] = {{}, {}, {}, {}};
#pragma unroll
    for (int mt = 0; mt < 4; ++mt) {
      v16h af = load_a_frag(ffW2h, 32, mt, 0, lr, l2);
      acc[mt] = __builtin_amdgcn_wmma_f32_16x16x32_f16(false, af, false, bf, (short)0, acc[mt], false, false);
    }
#pragma unroll
    for (int mt = 0; mt < 4; ++mt) {
      FragF ff; ff.v = acc[mt];
#pragma unroll
      for (int i = 0; i < 8; i += 2) {
        const int o = mt * 16 + 8 * l2 + i;
        const float v0 = 1.f / (1.f + __expf(-ff.f[i]));
        const float v1 = 1.f / (1.f + __expf(-ff.f[i + 1]));
        *(unsigned*)&buf[(p0 + lr) * SROW + o] = pack2(v0, v1);
      }
    }
  }
  __syncthreads();

  // ---------- Phase 4: gather + elementwise; stage e=[feat-x, x] (128 ch) ----------
  {
    const float* bxp = bx + ((size_t)b * 64) * N_ + nbr;
    const float* xp = x + ((size_t)b * 64) * N_ + n;
#pragma unroll
    for (int c = 0; c < 64; c += 2) {
      const float a0 = (float)buf[p * SROW + c];
      const float a1 = (float)buf[p * SROW + c + 1];
      const float g0 = bxp[(size_t)c * N_], g1 = bxp[(size_t)(c + 1) * N_];
      const float x0 = xp[(size_t)c * N_], x1 = xp[(size_t)(c + 1) * N_];
      const float f0 = fmaxf(prm[P_ACTG + c] * (rif[c] * a0 * g0) + prm[P_ACTB + c], 0.f);
      const float f1 = fmaxf(prm[P_ACTG + c + 1] * (rif[c + 1] * a1 * g1) + prm[P_ACTB + c + 1], 0.f);
      *(unsigned*)&buf[p * SROW + c] = pack2(f0 - x0, f1 - x1);
      *(unsigned*)&buf[p * SROW + 64 + c] = pack2(x0, x1);
    }
  }
  __syncthreads();

  // ---------- GEMM3 (WMMA): edge(64 x PIX) = edgeWh(64x128) * e; leaky; max over K ----------
  for (int nt = wv * 2; nt < wv * 2 + 2; ++nt) {
    const int p0 = nt * 16;
    v8f acc[4] = {{}, {}, {}, {}};
#pragma unroll
    for (int ks = 0; ks < 4; ++ks) {
      v16h bf = load_b_frag(buf, p0, ks * 32, lr, l2);
#pragma unroll
      for (int mt = 0; mt < 4; ++mt) {
        v16h af = load_a_frag(edgeWh, 128, mt, ks, lr, l2);
        acc[mt] = __builtin_amdgcn_wmma_f32_16x16x32_f16(false, af, false, bf, (short)0, acc[mt], false, false);
      }
    }
    const int pp = p0 + lr;
    const int ptl = pp / K_;
#pragma unroll
    for (int mt = 0; mt < 4; ++mt) {
      FragF ff; ff.v = acc[mt];
#pragma unroll
      for (int i = 0; i < 8; ++i) {
        const int o = mt * 16 + 8 * l2 + i;
        float v = ff.f[i] + prm[P_EDGEB + o];
        v = v > 0.f ? v : NEG_SLOPE * v;
        int iv = __float_as_int(v);
        iv = iv >= 0 ? iv : (iv ^ 0x7FFFFFFF);   // monotone int mapping for float max
        atomicMax(&outi[ptl * OUT_ + o], iv);
      }
    }
  }
  __syncthreads();

  for (int i = tid; i < NPTS * OUT_; i += PIX) {
    const int ptl = i >> 6, o = i & 63;
    int iv = outi[i];
    iv = iv >= 0 ? iv : (iv ^ 0x7FFFFFFF);
    out[((size_t)b * OUT_ + o) * N_ + n0 + ptl] = __int_as_float(iv);
  }
}

// ===================== launch =====================
extern "C" void kernel_launch(void* const* d_in, const int* in_sizes, int n_in,
                              void* d_out, int out_size, void* d_ws, size_t ws_size,
                              hipStream_t stream) {
  (void)in_sizes; (void)n_in; (void)out_size; (void)ws_size;
  const float* pos = (const float*)d_in[0];
  const float* x = (const float*)d_in[1];
  const float* glf = (const float*)d_in[2];
  const float* appf = (const float*)d_in[3];
  const float* basis_W = (const float*)d_in[4];
  const float* dk_W1 = (const float*)d_in[5];
  const float* dk_b1 = (const float*)d_in[6];
  const float* dk_g1 = (const float*)d_in[7];
  const float* dk_be1 = (const float*)d_in[8];
  const float* dk_W2 = (const float*)d_in[9];
  const float* dk_b2 = (const float*)d_in[10];
  const float* act_g = (const float*)d_in[11];
  const float* act_b = (const float*)d_in[12];
  const float* ff_W1 = (const float*)d_in[13];
  const float* ff_g1 = (const float*)d_in[14];
  const float* ff_be1 = (const float*)d_in[15];
  const float* ff_W2 = (const float*)d_in[16];
  const float* edge_W = (const float*)d_in[17];
  const float* edge_g = (const float*)d_in[18];
  const float* edge_be = (const float*)d_in[19];
  float* out = (float*)d_out;

  char* ws = (char*)d_ws;
  int* idx_ws = (int*)ws;                                   // 16*2048*20 ints  (2,621,440 B)
  float* bx_ws = (float*)(ws + 10485760);                   // 16*64*2048 f32   (8,388,608 B)
  f16* ffW1h = (f16*)(ws + 18874368);                       // 2048 halfs
  f16* ffW2h = ffW1h + 2048;                                // 2048 halfs
  f16* edgeWh = ffW2h + 2048;                               // 8192 halfs
  float* params = (float*)(ws + 18874368 + 24576);          // 912 floats

  knn_kernel<<<dim3(N_ / 256, B_), 256, 0, stream>>>(pos, idx_ws);
  bx_kernel<<<dim3(N_ / 64, B_), 256, 0, stream>>>(x, basis_W, bx_ws);
  prep_kernel<<<1, 512, 0, stream>>>(dk_W1, dk_b1, dk_g1, dk_be1, dk_W2, dk_b2,
                                     act_g, act_b, ff_W1, ff_g1, ff_be1, ff_W2,
                                     edge_W, edge_g, edge_be,
                                     ffW1h, ffW2h, edgeWh, params);
  fused_kernel<<<dim3(N_ / NPTS, B_), PIX, 0, stream>>>(x, glf, appf, idx_ws, bx_ws,
                                                        ffW1h, ffW2h, edgeWh, params, out);
}
